// Decoder_32676111188043
// MI455X (gfx1250) — matrix operand order repaired and checked by reading, not compiled, not run on hardware
//
#include <hip/hip_runtime.h>
#include <hip/hip_bf16.h>
#include <math.h>

// ---------------------------------------------------------------------------
// DCGRU decoder step on MI455X (gfx1250).
// Compute-bound (~0.5 TFLOP vs <0.5 GB traffic) -> everything heavy runs as
// v_wmma_f32_16x16x32_bf16 (K=32/issue, 8x the fp32 WMMA K-depth), fp32 accum.
// ---------------------------------------------------------------------------

#define BB 64
#define NN 1024
#define UU 128
#define DD 2
#define NM 5

typedef unsigned short u16t;
typedef unsigned int   u32t;
typedef __attribute__((ext_vector_type(16))) __bf16 v16bf;
typedef __attribute__((ext_vector_type(8)))  float  v8f;
typedef __attribute__((ext_vector_type(8)))  u32t   v8u;

// ---- bf16 helpers ---------------------------------------------------------
__device__ __forceinline__ u16t f2bf(float f) {
    union { float f; u32t u; } v; v.f = f;
    u32t u = v.u;
    u32t r = u + 0x7FFFu + ((u >> 16) & 1u);          // round-to-nearest-even
    if ((u & 0x7F800000u) == 0x7F800000u) r = u;      // inf/nan: truncate
    return (u16t)(r >> 16);
}
__device__ __forceinline__ float bf2f(u16t h) {
    union { u32t u; float f; } v; v.u = ((u32t)h) << 16;
    return v.f;
}

// ---- WMMA fragment gathers from LDS (layouts per CDNA5 ISA) ----
// LDS A tile: 128 rows x 32 cols bf16, row stride 40 bf16 (20 dwords: 16B-
// aligned for b128 staging stores, 16 distinct banks for fragment reads).
// A 16x32 frag: lane L -> M=L%16, half=L/16; VGPR p holds K pair
//   (p<4 ? 8*half+2p : 16+8*half+2(p-4)).
__device__ __forceinline__ v16bf load_afrag(const u32t* lAu, int mbase, int lane) {
    int M = lane & 15, half = lane >> 4;
    const u32t* p = lAu + (mbase + M) * 20;
    v8u r;
#pragma unroll
    for (int pi = 0; pi < 8; ++pi) {
        int K0 = (pi < 4) ? (8 * half + 2 * pi) : (16 + 8 * half + 2 * (pi - 4));
        r[pi] = p[K0 >> 1];
    }
    return __builtin_bit_cast(v16bf, r);
}
// LDS B tile: stored K-pair-transposed as [n(128)][kp(16)] dwords, stride 17
// (conflict-free). B 32x16 frag: lane L -> N=L%16, half=L/16; VGPR v
// holds dword kp = (v<4 ? 4*half + v : 8 + 4*half + (v-4)).
__device__ __forceinline__ v16bf load_bfrag(const u32t* lB, int nbase, int lane) {
    int Ncol = lane & 15, half = lane >> 4;
    const u32t* p = lB + (nbase + Ncol) * 17 + half * 4;
    v8u r;
#pragma unroll
    for (int v = 0; v < 8; ++v) r[v] = p[v < 4 ? v : v + 4];
    return __builtin_bit_cast(v16bf, r);
}

// ---- shared GEMM mainloop: C128x128 tile, 8 waves of 32x64, BK=32 ---------
// A: (rows x Kin) bf16 row-major per slab, slabs m = 0..nm-1 at slabA stride.
// B: (Kin x ldb)  bf16 row-major per slab at slabB stride.
template<bool BOUND>
__device__ __forceinline__ void gemm_mainloop(
    const u16t* __restrict__ Abase, long lda,
    const u16t* __restrict__ Bbase, long ldb,
    int Kin, int nm, long slabA, long slabB,
    int m0, int n0, v8f acc[2][4], u16t* lA, u32t* lB)
{
    const int t = threadIdx.x, lane = t & 31, w = t >> 5;
    const int wr = w >> 1, wc = w & 1;
    const int kTiles = (Kin + 31) >> 5;
    const int rA = t & 127, hA = t >> 7;   // bounded A staging: row / col-half
    const int nl = t & 127, kph = t >> 7;  // B staging: col / kp-half

    for (int m = 0; m < nm; ++m) {
        const u16t* A = Abase + (long)m * slabA;
        const u16t* B = Bbase + (long)m * slabB;
        for (int kt = 0; kt < kTiles; ++kt) {
            const int k0 = kt * 32;
            __syncthreads();
            // ---- stage A (128x32 bf16) ----
            if (!BOUND) {
#pragma unroll
                for (int it = 0; it < 2; ++it) {
                    int idx = t + it * 256;
                    int r = idx >> 2, q = idx & 3;
                    uint4 v = *(const uint4*)(A + (long)(m0 + r) * lda + k0 + q * 8);
                    *(uint4*)(&lA[r * 40 + q * 8]) = v;   // 16B-aligned (stride 80B)
                }
            } else {
#pragma unroll
                for (int c = 0; c < 16; ++c) {
                    int cc = hA * 16 + c;
                    int gk = k0 + cc;
                    u16t v = (gk < Kin) ? A[(long)(m0 + rA) * lda + gk] : (u16t)0;
                    lA[rA * 40 + cc] = v;
                }
            }
            // ---- stage B transposed into K-pairs: lB[n][kp] ----
#pragma unroll
            for (int i = 0; i < 8; ++i) {
                int kp = kph * 8 + i;
                int gk = k0 + 2 * kp;
                u32t lo, hi;
                if (!BOUND) {
                    lo = B[(long)gk * ldb + n0 + nl];
                    hi = B[(long)(gk + 1) * ldb + n0 + nl];
                } else {
                    lo = (gk     < Kin) ? (u32t)B[(long)gk * ldb + n0 + nl]       : 0u;
                    hi = (gk + 1 < Kin) ? (u32t)B[(long)(gk + 1) * ldb + n0 + nl] : 0u;
                }
                lB[nl * 17 + kp] = lo | (hi << 16);
            }
            __syncthreads();
            // ---- 8 WMMAs per wave per K-chunk ----
            const u32t* lAu = (const u32t*)lA;
            v16bf af0 = load_afrag(lAu, wr * 32,      lane);
            v16bf af1 = load_afrag(lAu, wr * 32 + 16, lane);
#pragma unroll
            for (int nf = 0; nf < 4; ++nf) {
                v16bf bf = load_bfrag(lB, wc * 64 + nf * 16, lane);
                acc[0][nf] = __builtin_amdgcn_wmma_f32_16x16x32_bf16(
                    false, af0, false, bf, (short)0, acc[0][nf], false, false);
                acc[1][nf] = __builtin_amdgcn_wmma_f32_16x16x32_bf16(
                    false, af1, false, bf, (short)0, acc[1][nf], false, false);
            }
        }
    }
}

// C/D 16x16 f32 layout: lane L -> N=L%16; VGPR r -> M = r + 8*(L/16).
#define EPILOGUE_LOOP(BODY)                                                    \
    {                                                                          \
        const int lane = threadIdx.x & 31, w = threadIdx.x >> 5;               \
        const int wr = w >> 1, wc = w & 1;                                     \
        const int Ncol = lane & 15, half = lane >> 4;                          \
        _Pragma("unroll") for (int i = 0; i < 2; ++i)                          \
        _Pragma("unroll") for (int nf = 0; nf < 4; ++nf)                       \
        _Pragma("unroll") for (int rr = 0; rr < 8; ++rr) {                     \
            long row = m0 + wr * 32 + i * 16 + rr + 8 * half;                  \
            long col = n0 + wc * 64 + nf * 16 + Ncol;                          \
            float v = acc[i][nf][rr];                                          \
            BODY                                                               \
        }                                                                      \
    }

// ---- diffusion GEMM: Y = S @ X   (or Y = 2*S@X - X0), bf16 out ------------
template<bool RESID>
__global__ void __launch_bounds__(256) diff_gemm_k(
    const u16t* __restrict__ S, const u16t* __restrict__ X,
    const u16t* __restrict__ X0, u16t* __restrict__ Y, int C)
{
    __shared__ u16t lA[128 * 40];
    __shared__ u32t lB[128 * 17];
    const int m0 = blockIdx.y * 128, n0 = blockIdx.x * 128;
    v8f acc[2][4];
#pragma unroll
    for (int i = 0; i < 2; ++i)
#pragma unroll
        for (int j = 0; j < 4; ++j) acc[i][j] = (v8f)(0.0f);
    gemm_mainloop<false>(S, NN, X, C, NN, 1, 0, 0, m0, n0, acc, lA, lB);
    EPILOGUE_LOOP({
        if (RESID) v = 2.0f * v - bf2f(X0[row * C + col]);
        Y[row * C + col] = f2bf(v);
    })
}

// ---- dense gate GEMM: sigmoid(sum_m XS_m @ Wg_m + b) -> r*h (bf16), u -----
template<bool BOUND>
__global__ void __launch_bounds__(256) dense_gate_k(
    const u16t* __restrict__ XS, long slabA,
    const u16t* __restrict__ W, long slabB, int in_size, int in_dim,
    const float* __restrict__ gb, const float* __restrict__ hold,
    float* __restrict__ ubuf, u16t* __restrict__ xc)
{
    __shared__ u16t lA[128 * 40];
    __shared__ u32t lB[128 * 17];
    const int m0 = blockIdx.y * 128, n0 = blockIdx.x * 128;
    v8f acc[2][4];
#pragma unroll
    for (int i = 0; i < 2; ++i)
#pragma unroll
        for (int j = 0; j < 4; ++j) acc[i][j] = (v8f)(0.0f);
    gemm_mainloop<BOUND>(XS, in_size, W, 2 * UU, in_size, NM, slabA, slabB,
                         m0, n0, acc, lA, lB);
    EPILOGUE_LOOP({
        int b = (int)(row & (BB - 1));
        long n = row >> 6;
        float s = 1.0f / (1.0f + __expf(-(v + gb[col])));
        if (col < UU) {   // r gate: write r*h into candidate X0 (bf16)
            float rh = s * hold[((long)b << 17) + (n << 7) + col];
            xc[row * in_size + in_dim + col] = f2bf(rh);
        } else {          // u gate
            ubuf[(row << 7) + (col - UU)] = s;
        }
    })
}

// ---- dense cand GEMM: tanh(...) + GRU combine -> h_new (f32, (B,N*U)) -----
template<bool BOUND>
__global__ void __launch_bounds__(256) dense_cand_k(
    const u16t* __restrict__ XS, long slabA,
    const u16t* __restrict__ W, long slabB, int in_size,
    const float* __restrict__ cb, const float* __restrict__ hold,
    const float* __restrict__ ubuf, float* __restrict__ hout)
{
    __shared__ u16t lA[128 * 40];
    __shared__ u32t lB[128 * 17];
    const int m0 = blockIdx.y * 128, n0 = blockIdx.x * 128;
    v8f acc[2][4];
#pragma unroll
    for (int i = 0; i < 2; ++i)
#pragma unroll
        for (int j = 0; j < 4; ++j) acc[i][j] = (v8f)(0.0f);
    gemm_mainloop<BOUND>(XS, in_size, W, UU, in_size, NM, slabA, slabB,
                         m0, n0, acc, lA, lB);
    EPILOGUE_LOOP({
        int b = (int)(row & (BB - 1));
        long n = row >> 6;
        float c = tanhf(v + cb[col]);
        float u = ubuf[(row << 7) + col];
        float ho = hold[((long)b << 17) + (n << 7) + col];
        hout[((long)b << 17) + (n << 7) + col] = u * ho + (1.0f - u) * c;
    })
}

// ---- support builders -----------------------------------------------------
__global__ void adj_sums_k(const float* __restrict__ adj,
                           float* __restrict__ rs, float* __restrict__ cs) {
    int i = blockIdx.x * blockDim.x + threadIdx.x;
    if (i >= NN) return;
    float r = 0.f, c = 0.f;
    for (int j = 0; j < NN; ++j) {
        r += adj[(long)i * NN + j];
        c += adj[(long)j * NN + i];
    }
    rs[i] = r; cs[i] = c;
}
// S0[i,j] = adj[j,i]/rowsum[j] ; S1[i,j] = adj[i,j]/colsum[j]  (both bf16)
__global__ void build_supports_k(const float* __restrict__ adj,
                                 const float* __restrict__ rs,
                                 const float* __restrict__ cs,
                                 u16t* __restrict__ s0, u16t* __restrict__ s1) {
    long idx = (long)blockIdx.x * blockDim.x + threadIdx.x;
    if (idx >= (long)NN * NN) return;
    int i = (int)(idx >> 10), j = (int)(idx & (NN - 1));
    s0[idx] = f2bf(adj[(long)j * NN + i] / rs[j]);
    s1[idx] = f2bf(adj[idx] / cs[j]);
}

// ---- weight repack: W[(c*NM+m), o] -> Wp[m][c][o] bf16 --------------------
__global__ void pack_w_k(const float* __restrict__ w, u16t* __restrict__ wp,
                         int in_size, int nout) {
    long total = (long)NM * in_size * nout;
    long idx = (long)blockIdx.x * blockDim.x + threadIdx.x;
    if (idx >= total) return;
    long m = idx / ((long)in_size * nout);
    long rem = idx % ((long)in_size * nout);
    long c = rem / nout;
    int  o = (int)(rem % nout);
    wp[idx] = f2bf(w[(c * NM + m) * nout + o]);
}

// ---- X0 pack: row = n*B+b, col = c; x part also copied into cand X0 -------
__global__ void pack_x0_k(const float* __restrict__ xsrc, int xdim,
                          const float* __restrict__ hsrc,
                          u16t* __restrict__ xg, u16t* __restrict__ xc,
                          int in_size) {
    long total = (long)BB * NN * in_size;
    long idx = (long)blockIdx.x * blockDim.x + threadIdx.x;
    if (idx >= total) return;
    long row = idx / in_size;
    int  c = (int)(idx % in_size);
    int  b = (int)(row & (BB - 1));
    long n = row >> 6;
    float v;
    if (c < xdim) {
        v = xsrc[((long)b * NN + n) * xdim + c];
        xc[idx] = f2bf(v);                 // x part shared with candidate X0
    } else {
        v = hsrc[((long)b << 17) + (n << 7) + (c - xdim)];
    }
    xg[idx] = f2bf(v);
}

// ---- final projection: out = h1 @ proj_w + proj_b  (K=128, N=2: VALU) -----
__global__ void proj_k(const float* __restrict__ h1, const float* __restrict__ pw,
                       const float* __restrict__ pb, float* __restrict__ out) {
    long r = (long)blockIdx.x * blockDim.x + threadIdx.x;   // r = b*N + n
    if (r >= (long)BB * NN) return;
    const float* h = h1 + (r << 7);
    float a0 = pb[0], a1 = pb[1];
#pragma unroll 8
    for (int u = 0; u < UU; ++u) {
        float x = h[u];
        a0 += x * pw[u * 2];
        a1 += x * pw[u * 2 + 1];
    }
    out[r * 2]     = a0;
    out[r * 2 + 1] = a1;
}

// ---------------------------------------------------------------------------
template<bool BOUND>
static void run_cell(const float* xsrc, int xdim, const float* hold,
                     const u16t* Wg, const u16t* Wc,
                     const float* gb, const float* cb, float* hout,
                     u16t* XS, u16t* XC, float* UB,
                     const u16t* S0, const u16t* S1, hipStream_t stream) {
    const int in_size = xdim + UU;
    const int C = BB * in_size;                 // diffusion GEMM N-dim
    const long slabA = (long)BB * NN * in_size; // elements per hop slab
    u16t* s0m = XS;
    u16t* s1m = XS + 1 * slabA;
    u16t* s2m = XS + 2 * slabA;
    u16t* s3m = XS + 3 * slabA;
    u16t* s4m = XS + 4 * slabA;
    const long tot = (long)BB * NN * in_size;

    auto diffuse = [&](u16t* x0) {
        dim3 gd(C / 128, NN / 128);
        diff_gemm_k<false><<<gd, 256, 0, stream>>>(S0, x0, nullptr, s1m, C);
        diff_gemm_k<false><<<gd, 256, 0, stream>>>(S1, x0, nullptr, s3m, C);
        diff_gemm_k<true ><<<gd, 256, 0, stream>>>(S0, s1m, x0, s2m, C);
        diff_gemm_k<true ><<<gd, 256, 0, stream>>>(S1, s3m, x0, s4m, C);
    };

    // gate pass
    pack_x0_k<<<(unsigned)((tot + 255) / 256), 256, 0, stream>>>(
        xsrc, xdim, hold, s0m, XC, in_size);
    diffuse(s0m);
    dense_gate_k<BOUND><<<dim3(2, 512), 256, 0, stream>>>(
        XS, slabA, Wg, (long)in_size * 2 * UU, in_size, xdim, gb, hold, UB, XC);

    // candidate pass: X0_cand (x part + r*h) becomes slab 0
    hipMemcpyAsync(s0m, XC, (size_t)slabA * sizeof(u16t),
                   hipMemcpyDeviceToDevice, stream);
    diffuse(s0m);
    dense_cand_k<BOUND><<<dim3(1, 512), 256, 0, stream>>>(
        XS, slabA, Wc, (long)in_size * UU, in_size, cb, hold, UB, hout);
}

extern "C" void kernel_launch(void* const* d_in, const int* in_sizes, int n_in,
                              void* d_out, int out_size, void* d_ws, size_t ws_size,
                              hipStream_t stream) {
    const float* inputs = (const float*)d_in[0];
    const float* hstate = (const float*)d_in[1];   // (2, B, N*U)
    const float* adj    = (const float*)d_in[2];
    const float* gw0    = (const float*)d_in[3];
    const float* gb0    = (const float*)d_in[4];
    const float* cw0    = (const float*)d_in[5];
    const float* cb0    = (const float*)d_in[6];
    const float* gw1    = (const float*)d_in[7];
    const float* gb1    = (const float*)d_in[8];
    const float* cw1    = (const float*)d_in[9];
    const float* cb1    = (const float*)d_in[10];
    const float* pw     = (const float*)d_in[11];
    const float* pb     = (const float*)d_in[12];

    float* out = (float*)d_out;                  // (B, N*D)
    float* h0  = out + (long)BB * NN * DD;       // (B, N*U)
    float* h1  = h0 + (long)BB * NN * UU;        // (B, N*U)
    const float* h0_old = hstate;
    const float* h1_old = hstate + (long)BB * NN * UU;

    // workspace carve-up (~238 MB)
    char* ws = (char*)d_ws;
    auto alloc = [&](size_t bytes) -> char* {
        char* p = ws;
        ws += (bytes + 255) & ~(size_t)255;
        return p;
    };
    u16t*  S0b = (u16t*)alloc((size_t)NN * NN * 2);
    u16t*  S1b = (u16t*)alloc((size_t)NN * NN * 2);
    float* rs  = (float*)alloc(NN * 4);
    float* cs  = (float*)alloc(NN * 4);
    u16t*  Wg0 = (u16t*)alloc((size_t)NM * 130 * 256 * 2);
    u16t*  Wc0 = (u16t*)alloc((size_t)NM * 130 * 128 * 2);
    u16t*  Wg1 = (u16t*)alloc((size_t)NM * 256 * 256 * 2);
    u16t*  Wc1 = (u16t*)alloc((size_t)NM * 256 * 128 * 2);
    u16t*  XS  = (u16t*)alloc((size_t)NM * BB * NN * 256 * 2);  // 5 hop slabs
    u16t*  XC  = (u16t*)alloc((size_t)BB * NN * 256 * 2);       // candidate X0
    float* UB  = (float*)alloc((size_t)BB * NN * UU * 4);       // u gate

    // supports (bf16) + packed weights (bf16)
    adj_sums_k<<<(NN + 255) / 256, 256, 0, stream>>>(adj, rs, cs);
    build_supports_k<<<(unsigned)(((long)NN * NN + 255) / 256), 256, 0, stream>>>(
        adj, rs, cs, S0b, S1b);
    pack_w_k<<<(unsigned)(((long)NM * 130 * 256 + 255) / 256), 256, 0, stream>>>(gw0, Wg0, 130, 256);
    pack_w_k<<<(unsigned)(((long)NM * 130 * 128 + 255) / 256), 256, 0, stream>>>(cw0, Wc0, 130, 128);
    pack_w_k<<<(unsigned)(((long)NM * 256 * 256 + 255) / 256), 256, 0, stream>>>(gw1, Wg1, 256, 256);
    pack_w_k<<<(unsigned)(((long)NM * 256 * 128 + 255) / 256), 256, 0, stream>>>(cw1, Wc1, 256, 128);

    // layer 0 (in_size=130 -> ragged K) and layer 1 (in_size=256 -> fast path)
    run_cell<true >(inputs, DD, h0_old, Wg0, Wc0, gb0, cb0, h0,
                    XS, XC, UB, S0b, S1b, stream);
    run_cell<false>(h0, UU, h1_old, Wg1, Wc1, gb1, cb1, h1,
                    XS, XC, UB, S0b, S1b, stream);

    // final projection
    proj_k<<<(BB * NN) / 256, 256, 0, stream>>>(h1, pw, pb, out);
}
